// Net9_14542759264805
// MI455X (gfx1250) — matrix operand order repaired and checked by reading, not compiled, not run on hardware
//
#include <hip/hip_runtime.h>
#include <hip/hip_bf16.h>
#include <math.h>

// ---------------------------------------------------------------------------
// CDNA5 / gfx1250 fused GNN forward.
// All dense math runs through v_wmma_f32_16x16x32_bf16 (wave32 WMMA).
// ---------------------------------------------------------------------------

typedef __attribute__((ext_vector_type(16))) __bf16 v16bf;
typedef __attribute__((ext_vector_type(8)))  float  v8f;
typedef __attribute__((ext_vector_type(4)))  unsigned int v4u;

union Frag {
    v16bf v;
    v4u   q[2];
};
static_assert(sizeof(Frag) == 32, "frag must be 32B");

struct __align__(8) bf16x4 { __bf16 x, y, z, w; };

__device__ __forceinline__ __bf16 f2bf(float x) {
    // round-to-nearest-even truncation, pure integer ops (always compiles)
    unsigned int u = __builtin_bit_cast(unsigned int, x);
    unsigned int r = (u + 0x7FFFu + ((u >> 16) & 1u)) >> 16;
    return __builtin_bit_cast(__bf16, (unsigned short)r);
}

// Segment of the (virtually concatenated / gathered) A operand.
struct SegDesc {
    const float* A;    // [rows, K] row-major
    const int*   idx;  // optional per-row gather (nullptr = identity)
    int          K;    // columns of this segment (multiple of 32)
};

#define BM 128
#define BN 128
#define BK 32
#define LDS_STRIDE 40   // padded bf16 K-stride: 80B rows keep 16B chunks aligned

// C[M,NN] = act( (concat(segs) @ W + bias) * (gscale*gmul) + gshift ) [+ resid]
__global__ __launch_bounds__(256)
void gemm_seg_kernel(SegDesc s0, SegDesc s1, SegDesc s2,
                     const float* __restrict__ W,      // [Ktot, NN] row-major
                     const float* __restrict__ bias,   // [NN] or null
                     const float* __restrict__ gscale, // [NN] or null (BN gamma)
                     float gmul,                       // 1/sqrt(1+eps)
                     const float* __restrict__ gshift, // [NN] or null (BN beta)
                     const float* __restrict__ resid,  // [M,NN] or null
                     float* __restrict__ C,
                     int M, int NN, int relu)
{
    __shared__ __align__(16) __bf16 lA[BM * LDS_STRIDE];
    __shared__ __align__(16) __bf16 lB[BN * LDS_STRIDE];

    const int tid  = threadIdx.x;
    const int lane = tid & 31;
    const int wave = tid >> 5;
    const int wm   = wave & 3;   // 4 waves along M (32 rows each)
    const int wn   = wave >> 2;  // 2 waves along N (64 cols each)
    const int h    = lane >> 4;  // lane half
    const int l16  = lane & 15;

    const int mBlock = blockIdx.y * BM;
    const int nBlock = blockIdx.x * BN;

    v8f acc[2][4];
#pragma unroll
    for (int mi = 0; mi < 2; ++mi)
#pragma unroll
        for (int ni = 0; ni < 4; ++ni)
            acc[mi][ni] = (v8f)0.0f;

    // staging coordinates (fixed per thread)
    const int am = tid >> 3;          // A row group base (0..31), rows am+32i
    const int akq = (tid & 7) * 4;    // A k quad (0,4,..,28)
    const int bk = tid >> 5;          // B k base (0..7), rows bk+8i
    const int bn4 = (tid & 31) * 4;   // B n quad (0..124)

    SegDesc segs[3] = {s0, s1, s2};
    int kbase = 0;                    // global K offset of current segment

    for (int sI = 0; sI < 3; ++sI) {
        const SegDesc s = segs[sI];
        if (s.K == 0) continue;

        // Resolve gathered row offsets ONCE per segment (4 rows per thread).
        int rowOff[4];
#pragma unroll
        for (int i = 0; i < 4; ++i) {
            int r = mBlock + am + 32 * i;
            if (s.idx) r = s.idx[r];
            rowOff[i] = r * s.K + akq;
        }

        for (int kl = 0; kl < s.K; kl += BK) {
            // ---- stage A tile: 4x global_load_b128 -> 4x ds_store_b64 ----
#pragma unroll
            for (int i = 0; i < 4; ++i) {
                const float4 v = *(const float4*)(s.A + (size_t)rowOff[i] + kl);
                bf16x4 p{f2bf(v.x), f2bf(v.y), f2bf(v.z), f2bf(v.w)};
                *(bf16x4*)&lA[(am + 32 * i) * LDS_STRIDE + akq] = p;
            }
            // ---- stage B tile transposed: lB[n][k] = W[kg+k][nBlock+n] ----
#pragma unroll
            for (int i = 0; i < 4; ++i) {
                const int k = bk + 8 * i;
                const float4 v = *(const float4*)(
                    W + (size_t)(kbase + kl + k) * NN + nBlock + bn4);
                lB[(bn4 + 0) * LDS_STRIDE + k] = f2bf(v.x);
                lB[(bn4 + 1) * LDS_STRIDE + k] = f2bf(v.y);
                lB[(bn4 + 2) * LDS_STRIDE + k] = f2bf(v.z);
                lB[(bn4 + 3) * LDS_STRIDE + k] = f2bf(v.w);
            }
            // prefetch next weight tile into GL2 while we compute
            __builtin_prefetch(W + (size_t)(kbase + kl + BK + bk) * NN + nBlock + bn4, 0, 3);

            __syncthreads();

            // ---- build WMMA fragments from LDS (2x ds_load_b128 each).
            // A 16x32 bf16: lane half h, M=lane%16; elems 0..7 -> K=8h+0..7,
            // elems 8..15 -> K=16+8h+0..7.  B 32x16: N=lane%16, same K order as A.
            const v4u* pA = (const v4u*)lA;
            const v4u* pB = (const v4u*)lB;

            Frag a[2], b[4];
#pragma unroll
            for (int mi = 0; mi < 2; ++mi) {
                int r = wm * 32 + mi * 16 + l16;
                a[mi].q[0] = pA[5 * r + h];        // bytes 80r + 16h
                a[mi].q[1] = pA[5 * r + 2 + h];    // bytes 80r + 32 + 16h
            }
#pragma unroll
            for (int ni = 0; ni < 4; ++ni) {
                int n = wn * 64 + ni * 16 + l16;
                b[ni].q[0] = pB[5 * n + h];
                b[ni].q[1] = pB[5 * n + 2 + h];
            }

#pragma unroll
            for (int mi = 0; mi < 2; ++mi)
#pragma unroll
                for (int ni = 0; ni < 4; ++ni)
                    acc[mi][ni] = __builtin_amdgcn_wmma_f32_16x16x32_bf16(
                        false, a[mi].v, false, b[ni].v,
                        (short)0, acc[mi][ni], false, false);

            __syncthreads();
        }
        kbase += s.K;
    }

    // ---- epilogue: bias, BN affine, ReLU, residual, store ----
#pragma unroll
    for (int mi = 0; mi < 2; ++mi) {
#pragma unroll
        for (int ni = 0; ni < 4; ++ni) {
#pragma unroll
            for (int r = 0; r < 8; ++r) {
                int row = mBlock + wm * 32 + mi * 16 + h * 8 + r;
                int col = nBlock + wn * 64 + ni * 16 + l16;
                float c = acc[mi][ni][r];
                if (bias)   c += bias[col];
                if (gscale) c = c * (gscale[col] * gmul) + gshift[col];
                if (relu)   c = fmaxf(c, 0.0f);
                if (resid)  c += resid[(size_t)row * NN + col];
                C[(size_t)row * NN + col] = c;
            }
        }
    }
}

// ---------------------------------------------------------------------------
// Attention: for each (b,p): s_q = <qv[b,q,:], img[b,p,:]>; a = softmax(1.4*s)
// over q; out[b,p,:] = xproj[b,p,:] + sum_q a_q * qp[b,q,:].
// One 256-thread block per global row (b*P+p).
// ---------------------------------------------------------------------------
__global__ __launch_bounds__(256)
void attn_kernel(const float* __restrict__ img,
                 const float* __restrict__ qv,
                 const float* __restrict__ qp,
                 const float* __restrict__ xproj,
                 float* __restrict__ out,
                 const int* __restrict__ num_props, int D, int Q)
{
    __shared__ float sh[8][257];
    const int P  = num_props[0];
    const int gp = blockIdx.x;          // 0 .. B*P-1
    const int b  = gp / P;
    const int tid = threadIdx.x;

    float part[8];
#pragma unroll
    for (int q = 0; q < 8; ++q) part[q] = 0.0f;

    for (int d = tid; d < D; d += 256) {
        float xi = img[(size_t)gp * D + d];
#pragma unroll
        for (int q = 0; q < 8; ++q)
            part[q] += xi * qv[(size_t)(b * Q + q) * D + d];
    }
#pragma unroll
    for (int q = 0; q < 8; ++q) sh[q][tid] = part[q];
    __syncthreads();
    for (int off = 128; off > 0; off >>= 1) {
        if (tid < off)
#pragma unroll
            for (int q = 0; q < 8; ++q) sh[q][tid] += sh[q][tid + off];
        __syncthreads();
    }

    float w[8];
    float mx = -3.4e38f;
#pragma unroll
    for (int q = 0; q < 8; ++q) mx = fmaxf(mx, 1.4f * sh[q][0]);
    float den = 0.0f;
#pragma unroll
    for (int q = 0; q < 8; ++q) { w[q] = __expf(1.4f * sh[q][0] - mx); den += w[q]; }
    float inv = 1.0f / den;

    for (int d = tid; d < D; d += 256) {
        float o = xproj[(size_t)gp * D + d];
#pragma unroll
        for (int q = 0; q < 8; ++q)
            o += (w[q] * inv) * qp[(size_t)(b * Q + q) * D + d];
        out[(size_t)gp * D + d] = o;
    }
}

// ---------------------------------------------------------------------------
// scatter-mean pieces
// ---------------------------------------------------------------------------
__global__ void scatter_msg_kernel(const float* __restrict__ msg,
                                   const int* __restrict__ col,
                                   float* __restrict__ agg, int E, int D)
{
    size_t i = (size_t)blockIdx.x * blockDim.x + threadIdx.x;
    if (i >= (size_t)E * D) return;
    int e = (int)(i / D);
    int d = (int)(i % D);
    atomicAdd(&agg[(size_t)col[e] * D + d], msg[i]);
}

__global__ void scatter_cnt_kernel(const int* __restrict__ col,
                                   float* __restrict__ cnt, int E)
{
    int e = blockIdx.x * blockDim.x + threadIdx.x;
    if (e < E) atomicAdd(&cnt[col[e]], 1.0f);
}

__global__ void div_cnt_kernel(float* __restrict__ agg,
                               const float* __restrict__ cnt, int N, int D)
{
    size_t i = (size_t)blockIdx.x * blockDim.x + threadIdx.x;
    if (i >= (size_t)N * D) return;
    agg[i] /= fmaxf(cnt[i / D], 1.0f);
}

// ---------------------------------------------------------------------------
// host side
// ---------------------------------------------------------------------------
static void gemm(hipStream_t st,
                 const float* A0, const int* i0, int K0,
                 const float* A1, const int* i1, int K1,
                 const float* A2, const int* i2, int K2,
                 const float* W, const float* bias,
                 const float* gscale, float gmul, const float* gshift,
                 const float* resid, float* C, int M, int NN, int relu)
{
    SegDesc s0{A0, i0, K0}, s1{A1, i1, K1}, s2{A2, i2, K2};
    dim3 grid(NN / BN, M / BM), block(256);
    gemm_seg_kernel<<<grid, block, 0, st>>>(s0, s1, s2, W, bias, gscale, gmul,
                                            gshift, resid, C, M, NN, relu);
}

extern "C" void kernel_launch(void* const* d_in, const int* in_sizes, int n_in,
                              void* d_out, int out_size, void* d_ws, size_t ws_size,
                              hipStream_t stream)
{
    const int D = 1024, Q = 8;
    const int Nn = in_sizes[0] / D;       // nodes (B*P) = 8192
    const int E  = in_sizes[2] / D;       // edges = 65536
    const int BQ = in_sizes[3] / D;       // B*Q = 128

    const float* x      = (const float*)d_in[0];
    const int*   eidx   = (const int*)d_in[1];     // [2,E]: row then col
    const float* eattr  = (const float*)d_in[2];
    const float* qry    = (const float*)d_in[3];
    const int*   nprops = (const int*)d_in[4];
    const int*   row = eidx;
    const int*   col = eidx + E;

    // params in setup_inputs() dict order, starting at d_in[5]
    const float* P_[32];
    for (int i = 0; i < 32; ++i) P_[i] = (const float*)d_in[5 + i];
    const float *t1w1=P_[0], *t1b1=P_[1], *t1g=P_[2],  *t1be=P_[3],
                *t1w2=P_[4], *t1b2=P_[5],
                *t2w1=P_[6], *t2b1=P_[7], *t2g=P_[8],  *t2be=P_[9],
                *t2w2=P_[10],*t2b2=P_[11],
                *qpw =P_[12],*qpb =P_[13],*pxw =P_[14],*pxb =P_[15],
                *mw1 =P_[16],*mb1 =P_[17],*mw2 =P_[18],*mb2 =P_[19],
                *ew1 =P_[20],*eb1 =P_[21],*ew2 =P_[22],*eb2 =P_[23],
                *n1w1=P_[24],*n1b1=P_[25],*n1w2=P_[26],*n1b2=P_[27],
                *n2w1=P_[28],*n2b1=P_[29],*n2w2=P_[30],*n2b2=P_[31];

    // workspace carve-out
    char* wsb = (char*)d_ws;
    size_t off = 0;
    auto carve = [&](size_t bytes) -> float* {
        float* p = (float*)(wsb + off);
        off = (off + bytes + 255) & ~(size_t)255;
        return p;
    };
    const size_t nd = (size_t)Nn * D * sizeof(float);
    const size_t ed = (size_t)E  * D * sizeof(float);
    const size_t qd = (size_t)BQ * D * sizeof(float);
    float* t1    = carve(nd);
    float* img   = carve(nd);
    float* qh    = carve(qd);
    float* qv    = carve(qd);
    float* qp    = carve(qd);
    float* xproj = carve(nd);
    float* xn    = carve(nd);
    float* mh    = carve(nd);
    float* xf    = carve(nd);
    float* big0  = carve(ed);   // edge hidden, then node1 hidden
    float* msg   = carve(ed);
    float* agg   = carve(nd);
    float* cnt   = carve((size_t)Nn * sizeof(float));
    float* n2h   = carve(nd);
    (void)ws_size; (void)n_in; (void)out_size;

    float* x_out = (float*)d_out;               // [N, D]
    float* e_out = (float*)d_out + (size_t)Nn * D;  // [E, D]

    const float gmul = 1.0f / sqrtf(1.0f + 1e-5f);

    // ---- theta1: img = lin2(relu(bn(lin1(xb)))) ----
    gemm(stream, x, nullptr, D, nullptr, nullptr, 0, nullptr, nullptr, 0,
         t1w1, t1b1, t1g, gmul, t1be, nullptr, t1, Nn, D, 1);
    gemm(stream, t1, nullptr, D, nullptr, nullptr, 0, nullptr, nullptr, 0,
         t1w2, t1b2, nullptr, 1.f, nullptr, nullptr, img, Nn, D, 0);
    // ---- theta2: qv ----
    gemm(stream, qry, nullptr, D, nullptr, nullptr, 0, nullptr, nullptr, 0,
         t2w1, t2b1, t2g, gmul, t2be, nullptr, qh, BQ, D, 1);
    gemm(stream, qh, nullptr, D, nullptr, nullptr, 0, nullptr, nullptr, 0,
         t2w2, t2b2, nullptr, 1.f, nullptr, nullptr, qv, BQ, D, 0);
    // ---- qp, xproj ----
    gemm(stream, qry, nullptr, D, nullptr, nullptr, 0, nullptr, nullptr, 0,
         qpw, qpb, nullptr, 1.f, nullptr, nullptr, qp, BQ, D, 0);
    gemm(stream, x, nullptr, D, nullptr, nullptr, 0, nullptr, nullptr, 0,
         pxw, pxb, nullptr, 1.f, nullptr, nullptr, xproj, Nn, D, 0);
    // ---- attention combine: xn = xproj + att . qp ----
    attn_kernel<<<dim3(Nn), dim3(256), 0, stream>>>(img, qv, qp, xproj, xn,
                                                    nprops, D, Q);
    // ---- node MLP: xf ----
    gemm(stream, xn, nullptr, D, nullptr, nullptr, 0, nullptr, nullptr, 0,
         mw1, mb1, nullptr, 1.f, nullptr, nullptr, mh, Nn, D, 1);
    gemm(stream, mh, nullptr, D, nullptr, nullptr, 0, nullptr, nullptr, 0,
         mw2, mb2, nullptr, 1.f, nullptr, nullptr, xf, Nn, D, 0);
    // ---- edge model: e = lin2(relu(lin1([xf[row], xf[col], eattr]))) ----
    gemm(stream, xf, row, D, xf, col, D, eattr, nullptr, D,
         ew1, eb1, nullptr, 1.f, nullptr, nullptr, big0, E, D, 1);
    gemm(stream, big0, nullptr, D, nullptr, nullptr, 0, nullptr, nullptr, 0,
         ew2, eb2, nullptr, 1.f, nullptr, nullptr, e_out, E, D, 0);
    // ---- node1 (messages): msg = lin2(relu(lin1([xf[row], e]))) ----
    gemm(stream, xf, row, D, e_out, nullptr, D, nullptr, nullptr, 0,
         n1w1, n1b1, nullptr, 1.f, nullptr, nullptr, big0, E, D, 1);
    gemm(stream, big0, nullptr, D, nullptr, nullptr, 0, nullptr, nullptr, 0,
         n1w2, n1b2, nullptr, 1.f, nullptr, nullptr, msg, E, D, 0);
    // ---- scatter-mean over col ----
    hipMemsetAsync(agg, 0, nd, stream);
    hipMemsetAsync(cnt, 0, (size_t)Nn * sizeof(float), stream);
    {
        size_t tot = (size_t)E * D;
        unsigned blocks = (unsigned)((tot + 255) / 256);
        scatter_msg_kernel<<<dim3(blocks), dim3(256), 0, stream>>>(msg, col, agg, E, D);
        scatter_cnt_kernel<<<dim3((E + 255) / 256), dim3(256), 0, stream>>>(col, cnt, E);
        size_t tn = (size_t)Nn * D;
        div_cnt_kernel<<<dim3((unsigned)((tn + 255) / 256)), dim3(256), 0, stream>>>(agg, cnt, Nn, D);
    }
    // ---- node2: x_out = lin2(relu(lin1([xf, agg]))) ----
    gemm(stream, xf, nullptr, D, agg, nullptr, D, nullptr, nullptr, 0,
         n2w1, n2b1, nullptr, 1.f, nullptr, nullptr, n2h, Nn, D, 1);
    gemm(stream, n2h, nullptr, D, nullptr, nullptr, 0, nullptr, nullptr, 0,
         n2w2, n2b2, nullptr, 1.f, nullptr, nullptr, x_out, Nn, D, 0);
}
